// Model_3221225472371
// MI455X (gfx1250) — matrix operand order repaired and checked by reading, not compiled, not run on hardware
//
#include <hip/hip_runtime.h>

typedef __attribute__((ext_vector_type(16))) _Float16 v16h;
typedef __attribute__((ext_vector_type(8)))  _Float16 v8h;
typedef __attribute__((ext_vector_type(8)))  float    v8f;

#define DIM 64

// ---------------------------------------------------------------- utilities
__global__ __launch_bounds__(256) void fill_f32(float* __restrict__ p, float v, int m) {
  int i = blockIdx.x * blockDim.x + threadIdx.x;
  if (i < m) p[i] = v;
}

__global__ __launch_bounds__(256) void deg_accum(const int* __restrict__ dst,
                                                 float* __restrict__ deg, int E) {
  int e = blockIdx.x * blockDim.x + threadIdx.x;
  if (e < E) unsafeAtomicAdd(&deg[dst[e]], 1.0f);   // hardware global_atomic_add_f32
}

__global__ __launch_bounds__(256) void rsqrt_inplace(float* __restrict__ p, int m) {
  int i = blockIdx.x * blockDim.x + threadIdx.x;
  if (i < m) p[i] = rsqrtf(p[i]);                   // deg >= 1 due to self-loops
}

// norm[e] = dinv[src]*dinv[dst] for real edges, dinv[i]^2 for self-loops
__global__ __launch_bounds__(256) void edge_norm(const int* __restrict__ src,
                                                 const int* __restrict__ dst,
                                                 const float* __restrict__ dinv,
                                                 float* __restrict__ nrm,
                                                 int E, int n) {
  int e = blockIdx.x * blockDim.x + threadIdx.x;
  if (e >= E + n) return;
  int s, t;
  if (e < E) { s = src[e]; t = dst[e]; } else { s = e - E; t = s; }
  nrm[e] = dinv[s] * dinv[t];
}

// h16 = emb[x] converted to f16 (GEMM input precision)
__global__ __launch_bounds__(256) void gather_emb(const int* __restrict__ x,
                                                  const float* __restrict__ emb,
                                                  _Float16* __restrict__ h16, int n) {
  int idx = blockIdx.x * blockDim.x + threadIdx.x;
  if (idx < n * DIM) {
    int i = idx >> 6, d = idx & 63;
    h16[idx] = (_Float16)emb[(size_t)x[i] * DIM + d];
  }
}

// ------------------------------------------------- [N,64] @ [64,64] via WMMA
// One wave computes a 16-row x 64-col f32 tile: 2 K-chunks x 4 col-tiles
// = 8 x v_wmma_f32_16x16x32_f16. W is pre-packed in LDS in the exact B
// fragment layout (v16h per lane -> 2x ds_load_b128); A is half a 128B f16 row
// per lane (4x global_load_b128). Full tiles take an unpredicated store path.
__global__ __launch_bounds__(256) void gemm64_wmma(const _Float16* __restrict__ H16,
                                                   const float* __restrict__ W,
                                                   float* __restrict__ O, int n) {
  __shared__ alignas(32) _Float16 Wb[8 * 32 * 16];   // [kc2*4+t][lane][16] = 8KB
  int tid = threadIdx.x;
  {
    int g = tid;                    // 0..255 <-> (kc2*4+t)*32 + lane
    int lane_ = g & 31;
    int kt = g >> 5;                // kc2*4 + t
    int kc2 = kt >> 2, t = kt & 3;
    int krow = kc2 * 32 + 16 * (lane_ >> 4);
    int col = t * 16 + (lane_ & 15);
    v16h tmp;
#pragma unroll
    for (int j = 0; j < 16; ++j)
      tmp[j] = (_Float16)W[(size_t)(krow + j) * DIM + col];
    *(v16h*)(Wb + (size_t)g * 16) = tmp;             // one 32B LDS store
  }
  __syncthreads();

  int lane = tid & 31;
  int gwave = blockIdx.x * 8 + (tid >> 5);
  int rowBase = gwave * 16;
  if (rowBase >= n) return;                          // wave-uniform: EXEC all-1s
  int lhi = lane >> 4;
  int llo = lane & 15;
  int arow = rowBase + llo;
  if (arow >= n) arow = n - 1;                       // clamp loads (ragged tail)

  v8f acc[4] = {};
#pragma unroll
  for (int kc2 = 0; kc2 < 2; ++kc2) {
    // A 16x32 f16: elem j -> K = kc + 16*lhi + j
    const _Float16* hrow = H16 + (size_t)arow * DIM + kc2 * 32 + lhi * 16;
    v8h alo = *(const v8h*)hrow;                     // 16B aligned
    v8h ahi = *(const v8h*)(hrow + 8);               // 16B aligned
    v16h a = __builtin_shufflevector(alo, ahi, 0, 1, 2, 3, 4, 5, 6, 7,
                                     8, 9, 10, 11, 12, 13, 14, 15);
#pragma unroll
    for (int t = 0; t < 4; ++t) {
      v16h b = *(const v16h*)(Wb + (size_t)((kc2 * 4 + t) * 32 + lane) * 16);
      acc[t] = __builtin_amdgcn_wmma_f32_16x16x32_f16(
          false, a, false, b, (short)0, acc[t], false, false);
    }
  }

  // C/D f32 layout: VGPR r -> M = r + 8*lhi, N = llo (per 16-col tile)
  int row0 = rowBase + lhi * 8;
  float* obase = O + (size_t)row0 * DIM + llo;
  if (rowBase + 16 <= n) {                           // wave-uniform fast path
#pragma unroll
    for (int r = 0; r < 8; ++r)
#pragma unroll
      for (int t = 0; t < 4; ++t)
        obase[(size_t)r * DIM + t * 16] = acc[t][r];
  } else {                                           // ragged last tile only
#pragma unroll
    for (int r = 0; r < 8; ++r)
      if (row0 + r < n)
#pragma unroll
        for (int t = 0; t < 4; ++t)
          obase[(size_t)r * DIM + t * 16] = acc[t][r];
  }
}

// accumulator init with self-loop term: OUT = HW * dinv^2  (no atomics needed)
__global__ __launch_bounds__(256) void self_loop_init(const float* __restrict__ HW,
                                                      const float* __restrict__ nrmSelf,
                                                      float* __restrict__ OUT, int n) {
  int idx = blockIdx.x * blockDim.x + threadIdx.x;
  if (idx < n * DIM) OUT[idx] = HW[idx] * nrmSelf[idx >> 6];
}

// ------------------------------------- edge scatter: out[dst] += hw[src]*norm
// 64 consecutive threads per edge -> coalesced row read + coalesced atomics.
__global__ __launch_bounds__(256) void scatter_msgs(const float* __restrict__ HW,
                                                    const int* __restrict__ src,
                                                    const int* __restrict__ dst,
                                                    const float* __restrict__ nrm,
                                                    float* __restrict__ OUT, int E) {
  int idx = blockIdx.x * blockDim.x + threadIdx.x;
  int total = E << 6;
  if (idx >= total) return;
  int e = idx >> 6, d = idx & 63;
  int s = src[e], t = dst[e];
  unsafeAtomicAdd(&OUT[(size_t)t * DIM + d], HW[(size_t)s * DIM + d] * nrm[e]);
}

// relu(in + b) -> f32 copy (pooling) and f16 copy (next GEMM's A operand)
__global__ __launch_bounds__(256) void bias_relu(const float* __restrict__ in,
                                                 const float* __restrict__ b,
                                                 float* __restrict__ out32,
                                                 _Float16* __restrict__ out16,
                                                 int total) {
  int idx = blockIdx.x * blockDim.x + threadIdx.x;
  if (idx < total) {
    float v = fmaxf(in[idx] + b[idx & 63], 0.0f);
    out32[idx] = v;
    out16[idx] = (_Float16)v;
  }
}

// ------------------------------------------------------------- mean pooling
__global__ __launch_bounds__(256) void pool_accum(const float* __restrict__ H,
                                                  const int* __restrict__ batch,
                                                  float* __restrict__ gsum,
                                                  int* __restrict__ gcnt, int n) {
  int idx = blockIdx.x * blockDim.x + threadIdx.x;
  if (idx >= n * DIM) return;
  int i = idx >> 6, d = idx & 63;
  int g = batch[i];
  unsafeAtomicAdd(&gsum[(size_t)g * DIM + d], H[idx]);
  if (d == 0) atomicAdd(&gcnt[g], 1);
}

__global__ __launch_bounds__(256) void final_mlp(const float* __restrict__ gsum,
                                                 const int* __restrict__ gcnt,
                                                 const float* __restrict__ Wc,
                                                 const float* __restrict__ bc,
                                                 float* __restrict__ out, int G) {
  int t = blockIdx.x * blockDim.x + threadIdx.x;
  if (t >= G * 2) return;
  int g = t >> 1, c = t & 1;
  float inv = 1.0f / fmaxf((float)gcnt[g], 1.0f);
  float acc = bc[c];
#pragma unroll
  for (int d = 0; d < DIM; ++d) acc += gsum[(size_t)g * DIM + d] * inv * Wc[d * 2 + c];
  out[t] = acc;
}

// ----------------------------------------------------------------- launcher
extern "C" void kernel_launch(void* const* d_in, const int* in_sizes, int n_in,
                              void* d_out, int out_size, void* d_ws, size_t ws_size,
                              hipStream_t stream) {
  const int*   x     = (const int*)d_in[0];
  const int*   ei    = (const int*)d_in[1];   // [2, E]: src row then dst row
  const int*   batch = (const int*)d_in[2];
  const float* emb   = (const float*)d_in[3];
  const float* W1    = (const float*)d_in[4];
  const float* b1    = (const float*)d_in[5];
  const float* W2    = (const float*)d_in[6];
  const float* b2    = (const float*)d_in[7];
  const float* Wc    = (const float*)d_in[8];
  const float* bc    = (const float*)d_in[9];
  (void)n_in; (void)ws_size;

  int n = in_sizes[0];
  int E = in_sizes[1] / 2;
  int G = out_size / 2;
  int nd = n * DIM;

  char* w = (char*)d_ws;
  auto take = [&](size_t bytes) {
    char* p = w;
    w += (bytes + 255) & ~(size_t)255;
    return p;
  };
  float*     dinv  = (float*)take((size_t)n * 4);
  float*     enorm = (float*)take((size_t)(E + n) * 4);
  _Float16*  h16   = (_Float16*)take((size_t)nd * 2);   // GEMM A operand (f16)
  float*     bufA  = (float*)take((size_t)nd * 4);      // relu output (f32, pooling)
  float*     bufB  = (float*)take((size_t)nd * 4);      // h @ W
  float*     bufC  = (float*)take((size_t)nd * 4);      // scatter accumulator
  float*     gsum  = (float*)take((size_t)G * DIM * 4);
  int*       gcnt  = (int*)take((size_t)G * 4);

  auto cdiv = [](int a, int b) { return (a + b - 1) / b; };

  // degree (with self-loop) -> dinv = rsqrt(deg) -> per-edge norms
  fill_f32<<<cdiv(n, 256), 256, 0, stream>>>(dinv, 1.0f, n);
  deg_accum<<<cdiv(E, 256), 256, 0, stream>>>(ei + E, dinv, E);
  rsqrt_inplace<<<cdiv(n, 256), 256, 0, stream>>>(dinv, n);
  edge_norm<<<cdiv(E + n, 256), 256, 0, stream>>>(ei, ei + E, dinv, enorm, E, n);

  // h16 = f16(emb[x])
  gather_emb<<<cdiv(nd, 256), 256, 0, stream>>>(x, emb, h16, n);

  int gemmBlocks = cdiv(cdiv(n, 16), 8);        // 8 waves / block, 16 rows / wave
  int scatTotal  = E * DIM;

  for (int layer = 0; layer < 2; ++layer) {
    const float* Wm = layer ? W2 : W1;
    const float* bm = layer ? b2 : b1;
    gemm64_wmma<<<gemmBlocks, 256, 0, stream>>>(h16, Wm, bufB, n);
    // accumulator starts at the self-loop contribution (pure stores, no atomics)
    self_loop_init<<<cdiv(nd, 256), 256, 0, stream>>>(bufB, enorm + E, bufC, n);
    scatter_msgs<<<cdiv(scatTotal, 256), 256, 0, stream>>>(bufB, ei, ei + E, enorm,
                                                           bufC, E);
    bias_relu<<<cdiv(nd, 256), 256, 0, stream>>>(bufC, bm, bufA, h16, nd);
  }

  // global mean pool + classifier head
  fill_f32<<<cdiv(G * DIM, 256), 256, 0, stream>>>(gsum, 0.0f, G * DIM);
  fill_f32<<<1, 256, 0, stream>>>((float*)gcnt, 0.0f, G);   // 0.0f bits == 0
  pool_accum<<<cdiv(nd, 256), 256, 0, stream>>>(bufA, batch, gsum, gcnt, n);
  final_mlp<<<1, 256, 0, stream>>>(gsum, gcnt, Wc, bc, (float*)d_out, G);
}
